// SdkBenchmarkHistogramTorusModel_3083786518604
// MI455X (gfx1250) — compile-verified, matrix operand order checked by reading, not run on hardware
//
#include <hip/hip_runtime.h>

// Problem constants (from reference)
#define HIST_TOTAL_BUCKETS 1024            // 8*8*16
#define BUCKET_SIZE 10
#define BATCHES 2
#define ELEMS_PER_BATCH (64 * 64 * 16)     // 65536
#define OUT_ELEMS (2 * HIST_TOTAL_BUCKETS) // 2048
// d_out layout: [0]=loss, [1..2048]=hist, [2049]=total_count, [2050]=max_abs_err

typedef __attribute__((ext_vector_type(16))) _Float16 v16h;
typedef __attribute__((ext_vector_type(8)))  float    v8f;

// ---------------------------------------------------------------------------
// Kernel 1: privatized LDS histogram, one 1024-thread block per batch.
// Bins land in d_ws as uint32 (plain stores, so no zeroing pass needed and the
// workspace is fully rewritten every call -> deterministic).
// ---------------------------------------------------------------------------
__global__ __launch_bounds__(1024) void hist_kernel(
    const long long* __restrict__ in, unsigned int* __restrict__ gbins) {
  __shared__ unsigned int hist[HIST_TOTAL_BUCKETS];
  const int tid = threadIdx.x;        // 0..1023
  const int batch = blockIdx.x;       // 0..1

  hist[tid] = 0u;                     // blockDim == HIST_TOTAL_BUCKETS
  __syncthreads();

  const long long* bin = in + (size_t)batch * ELEMS_PER_BATCH;
  constexpr int ITERS = ELEMS_PER_BATCH / 1024;  // 64
#pragma unroll 4
  for (int it = 0; it < ITERS; ++it) {
    const int idx = it * 1024 + tid;
    if (it + 1 < ITERS)
      __builtin_prefetch(&bin[idx + 1024], 0, 1);  // global_prefetch_b8
    const unsigned int v = (unsigned int)bin[idx]; // values < 10240 fit in u32
    const unsigned int bucket = v / BUCKET_SIZE;   // < 1024
    atomicAdd(&hist[bucket], 1u);                  // ds_add_u32
  }
  __syncthreads();

  gbins[batch * HIST_TOTAL_BUCKETS + tid] = hist[tid];
}

// ---------------------------------------------------------------------------
// Kernel 2: finalize. 256 threads compute hist->float, error^2 into LDS, and
// exact tree reductions for max|err| and total count. Wave 0 (full EXEC, as
// WMMA demands) then reduces the 2048 squared errors with
// v_wmma_f32_16x16x32_f16 against an all-ones B matrix: C[m][n] = sum_k A[m][k],
// so (column sum @ lane0) + (column sum @ lane16) == total sum of A.
// ---------------------------------------------------------------------------
__global__ __launch_bounds__(256) void finalize_kernel(
    const unsigned int* __restrict__ gbins, const float* __restrict__ ref,
    float* __restrict__ out) {
  __shared__ float sq[OUT_ELEMS];
  __shared__ float red[256];
  const int tid = threadIdx.x;

  float localMax = 0.0f;
  float localCnt = 0.0f;
#pragma unroll
  for (int i = tid; i < OUT_ELEMS; i += 256) {
    const float f = (float)gbins[i];
    out[1 + i] = f;                       // histogram output
    const float e = f - ref[i];
    sq[i] = e * e;
    localMax = fmaxf(localMax, fabsf(e));
    localCnt += f;
  }

  // max |error| reduction (exact, f32)
  red[tid] = localMax;
  __syncthreads();
  for (int s = 128; s > 0; s >>= 1) {
    if (tid < s) red[tid] = fmaxf(red[tid], red[tid + s]);
    __syncthreads();
  }
  if (tid == 0) out[2050] = red[0];
  __syncthreads();

  // total count reduction (exact: counts are small integers in f32)
  red[tid] = localCnt;
  __syncthreads();
  for (int s = 128; s > 0; s >>= 1) {
    if (tid < s) red[tid] += red[tid + s];
    __syncthreads();
  }
  if (tid == 0) out[2049] = red[0];
  __syncthreads();

  // Matrix-core reduction of sum(sq) on the first wave (32 lanes, EXEC all 1s).
  if (tid < 32) {
    v16h ones;
#pragma unroll
    for (int k = 0; k < 16; ++k) ones[k] = (_Float16)1.0f;

    v8f c = {};  // f32 accumulator
#pragma unroll
    for (int j = 0; j < 4; ++j) {        // 4 x 512 = 2048 values
      const float* p = &sq[j * 512 + tid * 16];
      v16h a;
#pragma unroll
      for (int k = 0; k < 16; ++k) a[k] = (_Float16)p[k];  // ds_load + cvt
      // 8 args: (neg_a, A, neg_b, B, c_mod, C, reuse_a, reuse_b)
      c = __builtin_amdgcn_wmma_f32_16x16x32_f16(
          false, a, false, ones, (short)0, c, false, false);
    }
    // Per-lane column sum: lanes 0-15 hold rows 0-7, lanes 16-31 rows 8-15.
    float s = c[0] + c[1] + c[2] + c[3] + c[4] + c[5] + c[6] + c[7];
    const float lo = __shfl(s, 0, 32);
    const float hi = __shfl(s, 16, 32);
    if (tid == 0) out[0] = (lo + hi) * (1.0f / (float)OUT_ELEMS);  // mean
  }
}

// ---------------------------------------------------------------------------
extern "C" void kernel_launch(void* const* d_in, const int* in_sizes, int n_in,
                              void* d_out, int out_size, void* d_ws, size_t ws_size,
                              hipStream_t stream) {
  (void)in_sizes; (void)n_in; (void)out_size; (void)ws_size;
  const long long* in  = (const long long*)d_in[0];  // int64 inputs (2,64,64,16)
  const float*     ref = (const float*)d_in[1];      // f32 ref_hist (2,8,8,16)
  float*           out = (float*)d_out;              // 2051 floats
  unsigned int*    ws  = (unsigned int*)d_ws;        // 2048 u32 bins

  hist_kernel<<<BATCHES, 1024, 0, stream>>>(in, ws);
  finalize_kernel<<<1, 256, 0, stream>>>(ws, ref, out);
}
